// GlobalFilter_27865747816687
// MI455X (gfx1250) — matrix operand order repaired and checked by reading, not compiled, not run on hardware
//
#include <hip/hip_runtime.h>
#include <hip/hip_bf16.h>
#include <hip/hip_fp16.h>

// GlobalFilter (FFT spectral filter) as two dense f16 WMMA GEMMs per batch.
// B=128, H=14, W=14, C=768.  F = 14*8*2 = 224 real-ified frequency rows.
//
//   Z[b]  (224x768) = A1 (224x196) @ x[b] (196x768)     [forward rfft2 basis]
//   Z2             = complex-weight * c_kw per (freq, channel)
//   y[b]  (196x768) = A1^T (196x224) @ Z2[b] (224x768)  [inverse basis]
//
// Memory-bound (~330 MB moved, ~14us at 23.3 TB/s); 17 GFLOP of f16 WMMA.
// Mainloop per 16x16 tile chunk: 2x global_load_b128 (A) + 2x ds_load_b128 (B)
// + 1x v_wmma_f32_16x16x32_f16.

typedef __attribute__((ext_vector_type(16))) _Float16 v16h;
typedef __attribute__((ext_vector_type(8)))  _Float16 v8h;
typedef __attribute__((ext_vector_type(4)))  _Float16 v4h;
typedef __attribute__((ext_vector_type(8)))  float    v8f;
typedef __attribute__((ext_vector_type(4)))  float    v4f;

#define BATCH   128
#define WW      14
#define CC      768
#define HWPIX   196          // 14*14
#define FREQ    224          // 14 * 8 * 2 (kh, kw, re/im)
#define KPAD    224          // 196 padded up to 7 * 32
#define MPAD2   208          // 196 padded up to 13 * 16
#define LDA     224          // row stride of A1 / A2 (f16 elements)
#define LDS_K   232          // padded k-stride of the LDS B panel (f16 elems)

// ---------------------------------------------------------------------------
// Build the real-ified rfft2 basis A1[f][k] and its transpose A2[k][f].
//   f = 2*(kh*8+kw) + {0:re, 1:im},  k = h*14 + w
//   A1[2q+0][k] =  cos(2*pi*(kh*h + kw*w)/14) / 14
//   A1[2q+1][k] = -sin(2*pi*(kh*h + kw*w)/14) / 14
// ---------------------------------------------------------------------------
__global__ void build_mats(_Float16* __restrict__ A1, _Float16* __restrict__ A2) {
    int idx = blockIdx.x * blockDim.x + threadIdx.x;
    if (idx >= FREQ * KPAD) return;
    int f = idx / KPAD;
    int k = idx % KPAD;
    float val = 0.0f;
    if (k < HWPIX) {
        int q  = f >> 1;
        int re = f & 1;
        int kh = q >> 3;
        int kw = q & 7;
        int h  = k / WW;
        int w  = k % WW;
        int t  = (kh * h + kw * w) % 14;
        float theta = (float)t * (6.28318530717958647692f / 14.0f);
        val = (re == 0 ? __cosf(theta) : -__sinf(theta)) * (1.0f / 14.0f);
    }
    A1[f * LDA + k] = (_Float16)val;
    if (k < MPAD2) A2[k * LDA + f] = (_Float16)val;   // rows 196..207 get zeros
}

// ---------------------------------------------------------------------------
// Convert x (f32, [B,196,768]) -> Xh (f16, [B,224,768]), 4 elems per thread.
// ---------------------------------------------------------------------------
__global__ void convert_x(const float* __restrict__ x, _Float16* __restrict__ Xh) {
    const long long total4 = (long long)BATCH * KPAD * CC / 4;
    long long idx = (long long)blockIdx.x * blockDim.x + threadIdx.x;
    if (idx >= total4) return;
    long long e = idx * 4;
    int c = (int)(e % CC);
    long long r = e / CC;
    int k = (int)(r % KPAD);
    int b = (int)(r / KPAD);
    v4h o = {};
    if (k < HWPIX) {
        v4f f = *(const v4f*)(x + ((long long)b * HWPIX + k) * CC + c);
        o[0] = (_Float16)f[0];
        o[1] = (_Float16)f[1];
        o[2] = (_Float16)f[2];
        o[3] = (_Float16)f[3];
    }
    *(v4h*)(Xh + e) = o;
}

// ---------------------------------------------------------------------------
// Cooperative stage of one 224x16 B panel into LDS, transposed to [n][k] so
// that each lane's WMMA B-operand runs (k = 8*half + e, 16 + 8*half + e) are contiguous.
// 256 threads, 1792 dword loads (2 f16 channels each), coalesced per k-row.
// ---------------------------------------------------------------------------
__device__ __forceinline__ void stage_B(const _Float16* __restrict__ Bm, int col0,
                                        _Float16 (*tileB)[LDS_K]) {
    const int t = threadIdx.x;
#pragma unroll
    for (int i = 0; i < 7; ++i) {
        const int widx = i * 256 + t;          // 0 .. 1791
        const int k  = widx >> 3;              // 0 .. 223
        const int wc = widx & 7;               // dword within the 16-col row
        unsigned d = *reinterpret_cast<const unsigned*>(
                         Bm + (long long)k * CC + col0 + wc * 2);
        union { unsigned u; _Float16 h[2]; } cv;
        cv.u = d;
        tileB[wc * 2 + 0][k] = cv.h[0];
        tileB[wc * 2 + 1][k] = cv.h[1];
    }
}

// ---------------------------------------------------------------------------
// One 16x16 tile: K over 224 in 7 chunks of 32 via v_wmma_f32_16x16x32_f16.
// A-operand (dense 16-bit A layout, ISA 7.12.2): lane r=lane%16, half=lane/16,
//   elements e0..7  -> k = 8*half + e          (16B contiguous run)
//   elements e8..15 -> k = 16 + 8*half + (e-8) (16B contiguous run)
// B-operand: lane n=lane%16, half=lane/16, same k order as A,
//   i.e. two 16B runs of tileB[n][...] -> two ds_load_b128.
// ---------------------------------------------------------------------------
__device__ __forceinline__ v8f wmma_tile_f16(const _Float16* __restrict__ A, int row0,
                                             const _Float16* ldsB) {
    const int lane = threadIdx.x & 31;
    const int r    = lane & 15;
    const int half = lane >> 4;
    const _Float16* arow = A + (long long)(row0 + r) * LDA;
    const _Float16* bcol = ldsB + r * LDS_K;
    v8f acc = {};
#pragma unroll
    for (int k0 = 0; k0 < KPAD; k0 += 32) {
        v8h a0 = *(const v8h*)(arow + k0 + 8 * half);
        v8h a1 = *(const v8h*)(arow + k0 + 16 + 8 * half);
        v8h b0 = *(const v8h*)(bcol + k0 + 8 * half);
        v8h b1 = *(const v8h*)(bcol + k0 + 16 + 8 * half);
        v16h av = __builtin_shufflevector(a0, a1,
                      0, 1, 2, 3, 4, 5, 6, 7, 8, 9, 10, 11, 12, 13, 14, 15);
        v16h bv = __builtin_shufflevector(b0, b1,
                      0, 1, 2, 3, 4, 5, 6, 7, 8, 9, 10, 11, 12, 13, 14, 15);
        acc = __builtin_amdgcn_wmma_f32_16x16x32_f16(
                  false, av, false, bv, (short)0, acc, false, false);
    }
    return acc;
}

// ---------------------------------------------------------------------------
// GEMM 1: Z = A1 @ X[b] with spectral-weight epilogue.  One block per
// (channel-tile, batch); 8 waves cover the 14 frequency tiles, sharing the
// LDS-staged X panel.  D-tile layout: elem i -> M = i + 8*half, N = lane%16,
// so (acc[2j], acc[2j+1]) are the (re, im) rows of one frequency.
// ---------------------------------------------------------------------------
__global__ __launch_bounds__(256)
void gemm1_fwd_weight(const _Float16* __restrict__ A1,
                      const _Float16* __restrict__ Xh,
                      const float*    __restrict__ cw,   // [14][8][768][2]
                      _Float16*       __restrict__ Z2) {
    __shared__ _Float16 tileB[16][LDS_K];
    const int nT   = blockIdx.x;            // 0..47
    const int b    = blockIdx.y;            // 0..127
    const int col0 = nT * 16;

    stage_B(Xh + (long long)b * KPAD * CC, col0, tileB);
    __syncthreads();

    const int wave = threadIdx.x >> 5;      // 0..7
    const int lane = threadIdx.x & 31;
    const int half = lane >> 4;
    const int ch   = col0 + (lane & 15);
    _Float16* Zb = Z2 + (long long)b * FREQ * CC;

    for (int mT = wave; mT < FREQ / 16; mT += 8) {
        if (mT + 8 < FREQ / 16)             // warm L2/WGP$ for the next panel
            __builtin_prefetch(A1 + (long long)((mT + 8) * 16 + (lane & 15)) * LDA, 0, 1);
        v8f acc = wmma_tile_f16(A1, mT * 16, &tileB[0][0]);
#pragma unroll
        for (int j = 0; j < 4; ++j) {
            const int f  = mT * 16 + 2 * j + 8 * half;   // even (re) row
            const int q  = f >> 1;
            const int kw = q & 7;
            const float* wp = cw + ((long long)q * CC + ch) * 2;
            const float wr = wp[0];
            const float wi = wp[1];
            const float s  = (kw == 0 || kw == 7) ? 1.0f : 2.0f;
            const float re = acc[2 * j];
            const float im = acc[2 * j + 1];
            Zb[(long long)(f + 0) * CC + ch] = (_Float16)((re * wr - im * wi) * s);
            Zb[(long long)(f + 1) * CC + ch] = (_Float16)((re * wi + im * wr) * s);
        }
    }
}

// ---------------------------------------------------------------------------
// GEMM 2: y[b] = A2 @ Z2[b] -> f32 d_out (rows < 196 only).
// ---------------------------------------------------------------------------
__global__ __launch_bounds__(256)
void gemm2_inv(const _Float16* __restrict__ A2,
               const _Float16* __restrict__ Z2,
               float*          __restrict__ out) {
    __shared__ _Float16 tileB[16][LDS_K];
    const int nT   = blockIdx.x;            // 0..47
    const int b    = blockIdx.y;            // 0..127
    const int col0 = nT * 16;

    stage_B(Z2 + (long long)b * FREQ * CC, col0, tileB);
    __syncthreads();

    const int wave = threadIdx.x >> 5;
    const int lane = threadIdx.x & 31;
    const int half = lane >> 4;
    const int ch   = col0 + (lane & 15);
    float* ob = out + (long long)b * HWPIX * CC;

    for (int mT = wave; mT < MPAD2 / 16; mT += 8) {
        if (mT + 8 < MPAD2 / 16)
            __builtin_prefetch(A2 + (long long)((mT + 8) * 16 + (lane & 15)) * LDA, 0, 1);
        v8f acc = wmma_tile_f16(A2, mT * 16, &tileB[0][0]);
#pragma unroll
        for (int i = 0; i < 8; ++i) {
            const int hw = mT * 16 + i + 8 * half;
            if (hw < HWPIX) ob[(long long)hw * CC + ch] = acc[i];
        }
    }
}

// ---------------------------------------------------------------------------
extern "C" void kernel_launch(void* const* d_in, const int* in_sizes, int n_in,
                              void* d_out, int out_size, void* d_ws, size_t ws_size,
                              hipStream_t stream) {
    const float* x  = (const float*)d_in[0];   // [128,14,14,768] f32
    const float* cw = (const float*)d_in[1];   // [14,8,768,2]    f32
    float* out      = (float*)d_out;           // [128,14,14,768] f32

    char* ws = (char*)d_ws;
    const size_t szA1 = (size_t)FREQ  * LDA * sizeof(_Float16);
    const size_t szA2 = (size_t)MPAD2 * LDA * sizeof(_Float16);
    const size_t offA1 = 0;
    const size_t offA2 = offA1 + szA1;
    const size_t offXh = (offA2 + szA2 + 255) & ~(size_t)255;
    const size_t szXh  = (size_t)BATCH * KPAD * CC * sizeof(_Float16); // 44 MB
    const size_t offZ2 = offXh + szXh;

    _Float16* A1 = (_Float16*)(ws + offA1);
    _Float16* A2 = (_Float16*)(ws + offA2);
    _Float16* Xh = (_Float16*)(ws + offXh);
    _Float16* Z2 = (_Float16*)(ws + offZ2);

    // 1) Build shared rfft2 basis matrices (tiny).
    build_mats<<<(FREQ * KPAD + 255) / 256, 256, 0, stream>>>(A1, A2);

    // 2) Stage x as f16 with zero K-padding (vectorized x4).
    const long long nconv4 = (long long)BATCH * KPAD * CC / 4;
    convert_x<<<(unsigned)((nconv4 + 255) / 256), 256, 0, stream>>>(x, Xh);

    // 3) Forward transform + spectral weight (WMMA, LDS-staged B panel).
    dim3 g1(CC / 16, BATCH);                 // 48 x 128 blocks, 8 waves each
    gemm1_fwd_weight<<<g1, 256, 0, stream>>>(A1, Xh, cw, Z2);

    // 4) Inverse transform (WMMA) -> d_out.
    dim3 g2(CC / 16, BATCH);
    gemm2_inv<<<g2, 256, 0, stream>>>(A2, Z2, out);
}
